// GIN_15633680957908
// MI455X (gfx1250) — matrix operand order repaired and checked by reading, not compiled, not run on hardware
//
#include <hip/hip_runtime.h>

// ---------- types ----------
typedef __attribute__((ext_vector_type(16))) __bf16 v16bf;
typedef __attribute__((ext_vector_type(8)))  float  v8f;
typedef unsigned int  u32x4 __attribute__((ext_vector_type(4)));
typedef unsigned int  u32x2 __attribute__((ext_vector_type(2)));
typedef float         f32x4 __attribute__((ext_vector_type(4)));

// exact pointee type the async-LDS builtin expects: int vector_size(16)
typedef int v4i_vs __attribute__((vector_size(16)));
typedef __attribute__((address_space(1))) v4i_vs* g_v4i;   // global
typedef __attribute__((address_space(3))) v4i_vs* l_v4i;   // LDS

union FragBF { u32x4 u[2]; v16bf v; };

#define D_FEAT 128
#define N_CLS  10

#if defined(__has_builtin)
#if __has_builtin(__builtin_amdgcn_global_load_async_to_lds_b128)
#define HAVE_ASYNC_LDS 1
#endif
#endif

__device__ __forceinline__ void wait_asynccnt0() {
#if defined(__has_builtin) && __has_builtin(__builtin_amdgcn_s_wait_asynccnt)
    __builtin_amdgcn_s_wait_asynccnt(0);
#else
    asm volatile("s_wait_asynccnt 0x0" ::: "memory");
#endif
}

__device__ __forceinline__ unsigned short f2bf(float f) {
    unsigned int u = __float_as_uint(f);
    unsigned int r = u + 0x7FFFu + ((u >> 16) & 1u);   // round-to-nearest-even
    return (unsigned short)(r >> 16);
}

// ---------- weight conversion ----------
__global__ __launch_bounds__(256) void convertbf_kernel(const float* __restrict__ s,
                                                        unsigned short* __restrict__ d, int n) {
    int i = blockIdx.x * 256 + threadIdx.x;
    if (i < n) d[i] = f2bf(s[i]);
}

// lin2_W [10,128] -> padded [16,128] bf16, rows 10..15 zero
__global__ __launch_bounds__(256) void pad_cls_kernel(const float* __restrict__ w,
                                                      unsigned short* __restrict__ d) {
    int i = blockIdx.x * 256 + threadIdx.x;
    if (i < 16 * D_FEAT) {
        int r = i >> 7;
        d[i] = (r < N_CLS) ? f2bf(w[i]) : (unsigned short)0;
    }
}

// ---------- edge scatter-add: agg[dst] += h[src] ----------
__global__ __launch_bounds__(256) void scatter_add_kernel(const float* __restrict__ h,
                                                          const long long* __restrict__ src,
                                                          const long long* __restrict__ dst,
                                                          float* __restrict__ agg, int nEdges) {
    int tid  = blockIdx.x * 256 + threadIdx.x;
    int e    = tid >> 5;          // one wave32 per edge
    int lane = tid & 31;          // each lane: 4 consecutive feats
    if (e >= nEdges) return;
    long long s = src[e];
    long long t = dst[e];
    f32x4 v = *reinterpret_cast<const f32x4*>(h + (size_t)s * D_FEAT + lane * 4);
    float* ap = agg + (size_t)t * D_FEAT + lane * 4;
    atomicAdd(ap + 0, v[0]);
    atomicAdd(ap + 1, v[1]);
    atomicAdd(ap + 2, v[2]);
    atomicAdd(ap + 3, v[3]);
}

// ---------- BN stats: per-column sum and sum-of-squares ----------
__global__ __launch_bounds__(256) void bn_stats_kernel(const float* __restrict__ z,
                                                       float* __restrict__ stats, int nRows) {
    int col = threadIdx.x & 127;
    int sub = threadIdx.x >> 7;                   // 0..1
    int r0   = blockIdx.x * 256 + sub;
    int rend = min(blockIdx.x * 256 + 256, nRows);
    float s = 0.f, s2 = 0.f;
    for (int r = r0; r < rend; r += 2) {
        float v = z[(size_t)r * D_FEAT + col];
        s += v; s2 += v * v;
    }
    atomicAdd(&stats[col], s);
    atomicAdd(&stats[D_FEAT + col], s2);
}

// ---------- BN apply + ReLU ----------
__global__ __launch_bounds__(256) void bn_apply_kernel(const float* __restrict__ z,
                                                       const float* __restrict__ stats,
                                                       const float* __restrict__ gamma,
                                                       const float* __restrict__ beta,
                                                       float* __restrict__ out, int nRows) {
    long long idx = (long long)blockIdx.x * 256 + threadIdx.x;
    if (idx >= (long long)nRows * D_FEAT) return;
    int col = (int)(idx & 127);
    float invN = 1.0f / (float)nRows;
    float mu   = stats[col] * invN;
    float var  = stats[D_FEAT + col] * invN - mu * mu;
    float sc   = gamma[col] * rsqrtf(var + 1e-5f);
    float sh   = beta[col] - mu * sc;
    float v = z[idx] * sc + sh;
    out[idx] = v > 0.f ? v : 0.f;
}

// ---------- LDS fill helper for bf16 weights (async-to-LDS on gfx1250) ----------
template<int N128>   // number of 16B chunks per thread
__device__ __forceinline__ void fill_lds_weights(unsigned short* lsW,
                                                 const unsigned short* __restrict__ Wb,
                                                 int t) {
#if defined(HAVE_ASYNC_LDS)
#pragma unroll
    for (int j = 0; j < N128; ++j) {
        int idx = t + j * 256;            // b128 chunk index
        __builtin_amdgcn_global_load_async_to_lds_b128(
            (g_v4i)(Wb + idx * 8),        // global src (AS1, const dropped)
            (l_v4i)(lsW + idx * 8),       // LDS dst (AS3)
            0, 0);
    }
    wait_asynccnt0();
#else
#pragma unroll
    for (int j = 0; j < N128; ++j) {
        int idx = t + j * 256;
        reinterpret_cast<u32x4*>(lsW)[idx] = reinterpret_cast<const u32x4*>(Wb)[idx];
    }
#endif
}

// ---------- main WMMA GEMM: out[M,128] = epi(A @ Wb^T + bias) ----------
// Wb stored row-major [out=128][in=128] bf16 (== B^T, i.e. Bt[n][k]).
// EPI: 0 = none (pre-BN z), 1 = ReLU.  FUSE: A = A0 + A1.
template<int EPI, bool FUSE>
__global__ __launch_bounds__(256) void gemm128_kernel(const float* __restrict__ A0,
                                                      const float* __restrict__ A1,
                                                      const unsigned short* __restrict__ Wb,
                                                      const float* __restrict__ bias,
                                                      float* __restrict__ out, int nRows) {
    __shared__ unsigned short lsA[D_FEAT * D_FEAT];   // 32 KB activation tile (bf16)
    __shared__ unsigned short lsW[D_FEAT * D_FEAT];   // 32 KB weight matrix (bf16)

    const int t         = threadIdx.x;
    const int blockBase = blockIdx.x * D_FEAT;

    // Weights: async DMA into LDS (ASYNCcnt path), completed before the A convert below.
    fill_lds_weights<8>(lsW, Wb, t);

    // Load + convert A tile (128 rows x 128 K), zero-pad tail rows.
#pragma unroll
    for (int j = 0; j < 16; ++j) {
        int e4   = t + j * 256;           // float4 index in tile
        int row  = e4 >> 5;               // 32 float4 per row
        int c4   = (e4 & 31) * 4;
        int grow = blockBase + row;
        f32x4 v = {0.f, 0.f, 0.f, 0.f};
        if (grow < nRows) {
            v = *reinterpret_cast<const f32x4*>(A0 + (size_t)grow * D_FEAT + c4);
            if (FUSE) {
                f32x4 w = *reinterpret_cast<const f32x4*>(A1 + (size_t)grow * D_FEAT + c4);
                v = v + w;
            }
        }
        u32x2 pk;
        pk[0] = (unsigned int)f2bf(v[0]) | ((unsigned int)f2bf(v[1]) << 16);
        pk[1] = (unsigned int)f2bf(v[2]) | ((unsigned int)f2bf(v[3]) << 16);
        *reinterpret_cast<u32x2*>(&lsA[row * D_FEAT + c4]) = pk;
    }
    __syncthreads();

    const int wave  = t >> 5;
    const int lane  = t & 31;
    const int lr    = lane & 15;
    const int half  = lane >> 4;
    const int mtile = wave;               // 8 waves x 16 rows = 128 rows

    v8f acc[8] = {};

#pragma unroll
    for (int ks = 0; ks < 4; ++ks) {
        // A fragment: lane (M=lr), K = ks*32 + 16*half + 0..15
        FragBF fa;
        int baseA = (mtile * 16 + lr) * D_FEAT + ks * 32 + half * 16;
        fa.u[0] = *reinterpret_cast<const u32x4*>(&lsA[baseA]);
        fa.u[1] = *reinterpret_cast<const u32x4*>(&lsA[baseA + 8]);
#pragma unroll
        for (int nt = 0; nt < 8; ++nt) {
            // B fragment: lane (N=lr), K = ks*32 + 16*half + 0..15
            FragBF fb;
            int baseB = (nt * 16 + lr) * D_FEAT + ks * 32 + half * 16;
            fb.u[0] = *reinterpret_cast<const u32x4*>(&lsW[baseB]);
            fb.u[1] = *reinterpret_cast<const u32x4*>(&lsW[baseB + 8]);
            acc[nt] = __builtin_amdgcn_wmma_f32_16x16x32_bf16(
                false, fa.v, false, fb.v, (short)0, acc[nt], false, false);
        }
    }

    // Epilogue + store (C/D layout: lane N=lr, VGPR v -> M = v + 8*half)
#pragma unroll
    for (int nt = 0; nt < 8; ++nt) {
        int col = nt * 16 + lr;
        float bv = bias[col];
#pragma unroll
        for (int v = 0; v < 8; ++v) {
            int grow = blockBase + mtile * 16 + half * 8 + v;
            if (grow < nRows) {
                float val = acc[nt][v] + bv;
                if (EPI == 1) val = fmaxf(val, 0.f);
                out[(size_t)grow * D_FEAT + col] = val;
            }
        }
    }
}

// ---------- classifier GEMM: out[M,10] = sigmoid(A @ Wb^T + bias), Wb padded [16,128] ----------
__global__ __launch_bounds__(256) void gemm_cls_kernel(const float* __restrict__ A0,
                                                       const unsigned short* __restrict__ Wb,
                                                       const float* __restrict__ bias,
                                                       float* __restrict__ out, int nRows) {
    __shared__ unsigned short lsA[D_FEAT * D_FEAT];   // 32 KB
    __shared__ unsigned short lsW[16 * D_FEAT];       // 4 KB

    const int t         = threadIdx.x;
    const int blockBase = blockIdx.x * D_FEAT;

    // 2048 ushorts = 256 x 16B: one async b128 per thread
    fill_lds_weights<1>(lsW, Wb, t);

#pragma unroll
    for (int j = 0; j < 16; ++j) {
        int e4   = t + j * 256;
        int row  = e4 >> 5;
        int c4   = (e4 & 31) * 4;
        int grow = blockBase + row;
        f32x4 v = {0.f, 0.f, 0.f, 0.f};
        if (grow < nRows)
            v = *reinterpret_cast<const f32x4*>(A0 + (size_t)grow * D_FEAT + c4);
        u32x2 pk;
        pk[0] = (unsigned int)f2bf(v[0]) | ((unsigned int)f2bf(v[1]) << 16);
        pk[1] = (unsigned int)f2bf(v[2]) | ((unsigned int)f2bf(v[3]) << 16);
        *reinterpret_cast<u32x2*>(&lsA[row * D_FEAT + c4]) = pk;
    }
    __syncthreads();

    const int wave  = t >> 5;
    const int lane  = t & 31;
    const int lr    = lane & 15;
    const int half  = lane >> 4;
    const int mtile = wave;

    v8f acc = {};
#pragma unroll
    for (int ks = 0; ks < 4; ++ks) {
        FragBF fa;
        int baseA = (mtile * 16 + lr) * D_FEAT + ks * 32 + half * 16;
        fa.u[0] = *reinterpret_cast<const u32x4*>(&lsA[baseA]);
        fa.u[1] = *reinterpret_cast<const u32x4*>(&lsA[baseA + 8]);
        FragBF fb;
        int baseB = lr * D_FEAT + ks * 32 + half * 16;
        fb.u[0] = *reinterpret_cast<const u32x4*>(&lsW[baseB]);
        fb.u[1] = *reinterpret_cast<const u32x4*>(&lsW[baseB + 8]);
        acc = __builtin_amdgcn_wmma_f32_16x16x32_bf16(
            false, fa.v, false, fb.v, (short)0, acc, false, false);
    }

    int col = lr;
    if (col < N_CLS) {
        float bv = bias[col];
#pragma unroll
        for (int v = 0; v < 8; ++v) {
            int grow = blockBase + mtile * 16 + half * 8 + v;
            if (grow < nRows) {
                float x = acc[v] + bv;
                // sigmoid with v_rcp_f32 instead of full IEEE divide
                out[(size_t)grow * N_CLS + col] =
                    __builtin_amdgcn_rcpf(1.0f + __expf(-x));
            }
        }
    }
}

// ---------- host orchestration ----------
extern "C" void kernel_launch(void* const* d_in, const int* in_sizes, int n_in,
                              void* d_out, int out_size, void* d_ws, size_t ws_size,
                              hipStream_t stream) {
    const float*     x     = (const float*)d_in[0];
    const long long* ei    = (const long long*)d_in[1];
    const float*     W1    = (const float*)d_in[2];
    const float*     b1    = (const float*)d_in[3];
    const float*     gamma = (const float*)d_in[4];
    const float*     beta  = (const float*)d_in[5];
    const float*     W2    = (const float*)d_in[6];
    const float*     b2    = (const float*)d_in[7];
    const float*     l1W   = (const float*)d_in[8];
    const float*     l1b   = (const float*)d_in[9];
    const float*     l2W   = (const float*)d_in[10];
    const float*     l2b   = (const float*)d_in[11];

    const int nNodes = in_sizes[0] / D_FEAT;
    const int nEdges = in_sizes[1] / 2;
    const long long* src = ei;
    const long long* dst = ei + nEdges;

    // ---- workspace carve-out ----
    char* ws = (char*)d_ws;
    unsigned short* W1b = (unsigned short*)ws;               // 3*16384 bf16
    unsigned short* W2b = W1b + 3 * 16384;                   // 3*16384 bf16
    unsigned short* L1b = W1b + 6 * 16384;                   // 16384 bf16
    unsigned short* L2b = W1b + 7 * 16384;                   // 2048 bf16 (padded)
    size_t woff = ((size_t)(7 * 16384 + 2048) * 2 + 255) & ~(size_t)255;
    float* stats = (float*)(ws + woff);  woff += 1024;
    size_t rowBytes = (size_t)nNodes * D_FEAT * sizeof(float);
    rowBytes = (rowBytes + 255) & ~(size_t)255;
    float* AGG = (float*)(ws + woff);  woff += rowBytes;     // also holds z (in-place)
    float* H1  = (float*)(ws + woff);  woff += rowBytes;
    float* H2  = (float*)(ws + woff);

    const int gemmBlocks  = (nNodes + 127) / 128;
    const int scatBlocks  = (nEdges * 32 + 255) / 256;
    const int statBlocks  = (nNodes + 255) / 256;
    const int elemBlocks  = (int)(((long long)nNodes * D_FEAT + 255) / 256);

    // ---- weights -> bf16 (tiny, once per launch) ----
    convertbf_kernel<<<(3 * 16384 + 255) / 256, 256, 0, stream>>>(W1, W1b, 3 * 16384);
    convertbf_kernel<<<(3 * 16384 + 255) / 256, 256, 0, stream>>>(W2, W2b, 3 * 16384);
    convertbf_kernel<<<(16384 + 255) / 256, 256, 0, stream>>>(l1W, L1b, 16384);
    pad_cls_kernel<<<8, 256, 0, stream>>>(l2W, L2b);

    const float* hin = x;
    for (int i = 0; i < 3; ++i) {
        // aggregation
        (void)hipMemsetAsync(AGG, 0, (size_t)nNodes * D_FEAT * sizeof(float), stream);
        scatter_add_kernel<<<scatBlocks, 256, 0, stream>>>(hin, src, dst, AGG, nEdges);
        // z = (h + agg) @ W1^T + b1   (z written in-place over AGG)
        gemm128_kernel<0, true><<<gemmBlocks, 256, 0, stream>>>(
            hin, AGG, W1b + (size_t)i * 16384, b1 + i * D_FEAT, AGG, nNodes);
        // batchnorm (training stats) + relu
        (void)hipMemsetAsync(stats, 0, 2 * D_FEAT * sizeof(float), stream);
        bn_stats_kernel<<<statBlocks, 256, 0, stream>>>(AGG, stats, nNodes);
        bn_apply_kernel<<<elemBlocks, 256, 0, stream>>>(AGG, stats,
            gamma + i * D_FEAT, beta + i * D_FEAT, H1, nNodes);
        // h = relu(z @ W2^T + b2)
        gemm128_kernel<1, false><<<gemmBlocks, 256, 0, stream>>>(
            H1, nullptr, W2b + (size_t)i * 16384, b2 + i * D_FEAT, H2, nNodes);
        hin = H2;
    }

    // head: relu(h @ lin1^T + b), then sigmoid(h @ lin2^T + b)
    gemm128_kernel<1, false><<<gemmBlocks, 256, 0, stream>>>(
        H2, nullptr, L1b, l1b, H1, nNodes);
    gemm_cls_kernel<<<gemmBlocks, 256, 0, stream>>>(
        H1, L2b, l2b, (float*)d_out, nNodes);
}
